// Attention_78898549227880
// MI455X (gfx1250) — compile-verified, matrix operand order checked by reading, not run on hardware
//
#include <hip/hip_runtime.h>
#include <hip/hip_bf16.h>
#include <math.h>

typedef __attribute__((ext_vector_type(16))) _Float16 v16h;
typedef __attribute__((ext_vector_type(8)))  float    v8f;
typedef __attribute__((ext_vector_type(8)))  unsigned int v8u;
typedef __attribute__((ext_vector_type(4)))  float    v4f;

namespace {

constexpr int kNB   = 4;
constexpr int kNN   = 8192;
constexpr int kKK   = 32;     // keys per position (== wave32 lane count!)
constexpr int kDV   = 256;
constexpr int kHID  = 64;
constexpr int WAVES = 4;      // positions per block (1 wave each)
constexpr int HS    = 72;     // LDS activation row stride in halves (16B-aligned + bank pad)
constexpr int WS    = 72;     // LDS weight row stride in halves
constexpr int HROWS = 34;     // 32 k-rows + q row (32) + scratch row (33)
constexpr float kScale = 0.57735026918962576f;  // 1/sqrt(3), pre-MLP q dim

union HalfPair { _Float16 f[2]; unsigned int u; };
union Frag     { v8u u; v16h h; };

// Gather one 16x32 f16 WMMA fragment from an LDS row-major matrix.
// ISA 16-bit A layout: lane<16 holds K {0..7,16..23}, lane>=16 holds {8..15,24..31},
// two packed halves per VGPR. Same gather works for B (row = output feature n).
__device__ __forceinline__ v16h load_frag(const _Float16* base, int row, int stride,
                                          int cbase, int hi) {
  Frag f;
#pragma unroll
  for (int vv = 0; vv < 8; ++vv) {
    int k0 = cbase + ((vv & 4) ? 16 : 0) + hi * 8 + (vv & 3) * 2;
    f.u[vv] = *(const unsigned int*)(base + row * stride + k0);
  }
  return f.h;
}

// One 64->64 layer on the [32x64] k tile AND the q row via WMMA f32<=f16.
// Rowgroups 0,1: k rows, weights wK/bK. Rowgroup 2: q row broadcast into all
// 16 A rows (uniform LDS read), weights wQ/bQ; only D VGPR0 is kept (lanes>=16
// write scratch row 33 so control flow stays uniform around the WMMAs).
template <bool RELU>
__device__ __forceinline__ void mlp_layer_wmma(_Float16* h,
                                               const _Float16* wK, const _Float16* wQ,
                                               const float* __restrict__ bK,
                                               const float* __restrict__ bQ,
                                               int lh, int hi) {
#pragma unroll
  for (int g = 0; g < 3; ++g) {
    const int arow = (g < 2) ? (g * 16 + lh) : 32;
    const _Float16* w  = (g < 2) ? wK : wQ;
    const float* bias  = (g < 2) ? bK : bQ;
    v16h a0 = load_frag(h, arow, HS, 0,  hi);
    v16h a1 = load_frag(h, arow, HS, 32, hi);
#pragma unroll
    for (int t = 0; t < 4; ++t) {
      const int n = t * 16 + lh;
      const float bn = bias[n];
      v8f acc;
#pragma unroll
      for (int i = 0; i < 8; ++i) acc[i] = bn;
      v16h fb0 = load_frag(w, n, WS, 0,  hi);
      v16h fb1 = load_frag(w, n, WS, 32, hi);
      acc = __builtin_amdgcn_wmma_f32_16x16x32_f16(false, a0, false, fb0,
                                                   (short)0, acc, false, false);
      acc = __builtin_amdgcn_wmma_f32_16x16x32_f16(false, a1, false, fb1,
                                                   (short)0, acc, false, false);
      if (g < 2) {
#pragma unroll
        for (int i = 0; i < 8; ++i) {
          float r = acc[i];
          if (RELU) r = fmaxf(r, 0.0f);
          // D layout: VGPR i -> M = i (+8 for upper half-wave)
          h[(g * 16 + hi * 8 + i) * HS + n] = (_Float16)r;
        }
      } else {
        float r = acc[0];
        if (RELU) r = fmaxf(r, 0.0f);
        h[(32 + hi) * HS + n] = (_Float16)r;  // hi==1 lanes hit scratch row 33
      }
    }
  }
}

__global__ __launch_bounds__(WAVES * 32, 1) void fused_mlp_attn(
    const float* __restrict__ q,   const float* __restrict__ kin,
    const float* __restrict__ vin,
    const float* __restrict__ qW0, const float* __restrict__ qB0,
    const float* __restrict__ qW1, const float* __restrict__ qB1,
    const float* __restrict__ qW2, const float* __restrict__ qB2,
    const float* __restrict__ kW0, const float* __restrict__ kB0,
    const float* __restrict__ kW1, const float* __restrict__ kB1,
    const float* __restrict__ kW2, const float* __restrict__ kB2,
    float* __restrict__ out) {
  __shared__ _Float16 sWk1[kHID * WS];          // k layer1 weights, f16
  __shared__ _Float16 sWk2[kHID * WS];          // k layer2 weights, f16
  __shared__ _Float16 sWq1[kHID * WS];          // q layer1 weights, f16
  __shared__ _Float16 sWq2[kHID * WS];          // q layer2 weights, f16
  __shared__ float    sW0[kHID * 4];            // k layer0 weights (stride 4)
  __shared__ float    sB0[kHID];
  __shared__ _Float16 sH[WAVES][HROWS * HS];    // per-wave activation tile

  const int tid = threadIdx.x;

  // Stage shared weights once per block.
  for (int i = tid; i < kHID * kHID; i += WAVES * 32) {
    const int r = (i >> 6) * WS + (i & 63);
    sWk1[r] = (_Float16)kW1[i];
    sWk2[r] = (_Float16)kW2[i];
    sWq1[r] = (_Float16)qW1[i];
    sWq2[r] = (_Float16)qW2[i];
  }
  if (tid < kHID) {
    sW0[tid * 4 + 0] = kW0[tid * 3 + 0];
    sW0[tid * 4 + 1] = kW0[tid * 3 + 1];
    sW0[tid * 4 + 2] = kW0[tid * 3 + 2];
    sB0[tid] = kB0[tid];
  }
  __syncthreads();

  const int wave = tid >> 5;
  const int lane = tid & 31;
  const int lh   = lane & 15;
  const int hi   = lane >> 4;
  const long pos = (long)blockIdx.x * WAVES + wave;

  const float* kp = kin + pos * (kKK * 3);
  const float* vp = vin + pos * (kKK * kDV);
  const float* qp = q   + pos * 3;
  _Float16* h = sH[wave];

  // Prefetch the entire 32KB v tile (the bandwidth roofline term) so HBM
  // latency overlaps the WMMA MLP phase. 8 x 32 lanes x 128B = 32KB.
#pragma unroll
  for (int i = 0; i < 8; ++i)
    __builtin_prefetch(vp + (i * 32 + lane) * 32, 0, 1);

  // ---- layer 0 (3 -> 64), VALU; one k row per lane (K == 32 == wave32) ----
  {
    const float x0 = kp[lane * 3 + 0];
    const float x1 = kp[lane * 3 + 1];
    const float x2 = kp[lane * 3 + 2];
#pragma unroll
    for (int j = 0; j < kHID; j += 2) {
      HalfPair p;
#pragma unroll
      for (int s = 0; s < 2; ++s) {
        float a = fmaf(x2, sW0[(j + s) * 4 + 2],
                  fmaf(x1, sW0[(j + s) * 4 + 1],
                  fmaf(x0, sW0[(j + s) * 4 + 0], sB0[j + s])));
        p.f[s] = (_Float16)fmaxf(a, 0.0f);
      }
      *(unsigned int*)(h + lane * HS + j) = p.u;
    }
    // q layer 0 -> activation row 32: lane computes features 2*lane, 2*lane+1
    const float y0 = qp[0], y1 = qp[1], y2 = qp[2];
    HalfPair p;
#pragma unroll
    for (int s = 0; s < 2; ++s) {
      const int j = 2 * lane + s;
      float a = fmaf(y2, qW0[j * 3 + 2],
                fmaf(y1, qW0[j * 3 + 1],
                fmaf(y0, qW0[j * 3 + 0], qB0[j])));
      p.f[s] = (_Float16)fmaxf(a, 0.0f);
    }
    *(unsigned int*)(h + 32 * HS + 2 * lane) = p.u;
  }
  __syncthreads();

  // ---- layer 1 (64 -> 64), WMMA, fused bias + ReLU, k rows + q row ----
  mlp_layer_wmma<true>(h, sWk1, sWq1, kB1, qB1, lh, hi);
  __syncthreads();

  // ---- layer 2 (64 -> 64), WMMA, fused bias, linear ----
  mlp_layer_wmma<false>(h, sWk2, sWq2, kB2, qB2, lh, hi);
  __syncthreads();

  // ---- logits (lane r owns key r) + softmax across the wave ----
  float lg = 0.0f;
#pragma unroll 8
  for (int c = 0; c < kHID; ++c)
    lg = fmaf((float)h[lane * HS + c], (float)h[32 * HS + c], lg);
  const float x = lg * kScale;
  float mx = x;
#pragma unroll
  for (int off = 16; off > 0; off >>= 1) mx = fmaxf(mx, __shfl_xor(mx, off, 32));
  const float e = __expf(x - mx);
  float s = e;
#pragma unroll
  for (int off = 16; off > 0; off >>= 1) s += __shfl_xor(s, off, 32);
  const float wgt = e / s;

  // ---- bandwidth-critical: stream v[32][256], non-temporal (read-once) ----
  float a0 = 0, a1 = 0, a2 = 0, a3 = 0, c0 = 0, c1 = 0, c2 = 0, c3 = 0;
#pragma unroll 4
  for (int r = 0; r < kKK; ++r) {
    const float wr = __shfl(wgt, r, 32);
    const v4f* vr = (const v4f*)(vp + r * kDV);
    const v4f u = __builtin_nontemporal_load(vr + lane);        // d = 4*lane..
    const v4f t = __builtin_nontemporal_load(vr + lane + 32);   // d = 128+4*lane..
    a0 = fmaf(wr, u.x, a0); a1 = fmaf(wr, u.y, a1);
    a2 = fmaf(wr, u.z, a2); a3 = fmaf(wr, u.w, a3);
    c0 = fmaf(wr, t.x, c0); c1 = fmaf(wr, t.y, c1);
    c2 = fmaf(wr, t.z, c2); c3 = fmaf(wr, t.w, c3);
  }
  v4f r0 = {a0, a1, a2, a3};
  v4f r1 = {c0, c1, c2, c3};
  v4f* op = (v4f*)(out + pos * kDV);
  __builtin_nontemporal_store(r0, op + lane);
  __builtin_nontemporal_store(r1, op + lane + 32);
}

}  // namespace

extern "C" void kernel_launch(void* const* d_in, const int* in_sizes, int n_in,
                              void* d_out, int out_size, void* d_ws, size_t ws_size,
                              hipStream_t stream) {
  const float* q   = (const float*)d_in[0];
  const float* k   = (const float*)d_in[1];
  const float* v   = (const float*)d_in[2];
  const float* qW0 = (const float*)d_in[3];
  const float* qB0 = (const float*)d_in[4];
  const float* qW1 = (const float*)d_in[5];
  const float* qB1 = (const float*)d_in[6];
  const float* qW2 = (const float*)d_in[7];
  const float* qB2 = (const float*)d_in[8];
  const float* kW0 = (const float*)d_in[9];
  const float* kB0 = (const float*)d_in[10];
  const float* kW1 = (const float*)d_in[11];
  const float* kB1 = (const float*)d_in[12];
  const float* kW2 = (const float*)d_in[13];
  const float* kB2 = (const float*)d_in[14];
  float* out = (float*)d_out;

  const int positions = kNB * kNN;          // 32768
  const int blocks = positions / WAVES;     // 8192
  fused_mlp_attn<<<blocks, WAVES * 32, 0, stream>>>(
      q, k, v, qW0, qB0, qW1, qB1, qW2, qB2,
      kW0, kB0, kW1, kB1, kW2, kB2, out);
}
